// Attention_1529008357672
// MI455X (gfx1250) — compile-verified, matrix operand order checked by reading, not run on hardware
//
#include <hip/hip_runtime.h>
#include <hip/hip_bf16.h>

// ---------------------------------------------------------------------------
// Attention forward for B=2, S=2048, DIM=2048, H=16, HD=128 on gfx1250.
// bf16 WMMA (v_wmma_f32_16x16x32_bf16) for all matmuls; LDS-tiled GEMM with
// async global->LDS staging (ASYNCcnt) when available; flash-attention for
// the softmax stage. fp32 accumulate everywhere.
// ---------------------------------------------------------------------------

typedef __attribute__((ext_vector_type(16))) __bf16 v16bf;
typedef __attribute__((ext_vector_type(8)))  __bf16 bf16x8;
typedef __attribute__((ext_vector_type(8)))  float  v8f;
typedef __attribute__((ext_vector_type(4)))  int    v4i;

#define B_    2
#define S_    2048
#define DIM_  2048
#define H_    16
#define HD_   128
#define MROWS (B_ * S_)   // 4096

#if defined(__has_builtin)
#if __has_builtin(__builtin_amdgcn_global_load_async_to_lds_b128) && \
    __has_builtin(__builtin_amdgcn_s_wait_asynccnt)
#define USE_ASYNC_LDS 1
#endif
#endif
#ifndef USE_ASYNC_LDS
#define USE_ASYNC_LDS 0
#endif

// Per-lane fragment load for WMMA A/B operands (bf16, 16x16x32 shape).
// Documented layout: element i of the 16-elem fragment maps to
//   k = (i/8)*16 + (lane/16)*8 + (i%8)
// so each lane needs 8 contiguous bf16 at (k-base + half*8) and 8 more at +16.
// `base` must already point at (row_ptr + kbase + half*8).
static __device__ __forceinline__ v16bf load_frag(const __bf16* base) {
  bf16x8 lo = *(const bf16x8*)(base);
  bf16x8 hi = *(const bf16x8*)(base + 16);
  v16bf r;
#pragma unroll
  for (int i = 0; i < 8; ++i) { r[i] = lo[i]; r[8 + i] = hi[i]; }
  return r;
}

static __device__ __forceinline__ v8f wmma_bf16(v16bf a, v16bf b, v8f c) {
  return __builtin_amdgcn_wmma_f32_16x16x32_bf16(
      false, a, false, b, (short)0, c, false, false);
}

// ---------------------------------------------------------------------------
// fp32 -> bf16 elementwise conversion
// ---------------------------------------------------------------------------
__global__ void f32_to_bf16_kernel(const float* __restrict__ in,
                                   __bf16* __restrict__ out, int n) {
  int i = blockIdx.x * blockDim.x + threadIdx.x;
  if (i < n) out[i] = (__bf16)in[i];
}

// ---------------------------------------------------------------------------
// C[M,N] = A[M,K](bf16) @ W[N,K](bf16)^T + bias[N]   (torch Linear semantics)
//
// Block = 256 threads (8 waves) -> 128x128 output tile.
// Waves arranged 4(m) x 2(n); each wave computes 32x64 = 8 WMMA tiles/k-step.
// A/W k-slices (128 rows x 32 k) staged in double-buffered LDS, rows padded
// to 40 elements to avoid bank conflicts on fragment reads.
// Staging uses global_load_async_to_lds_b128 when the toolchain exposes it
// (ASYNCcnt + barrier pipeline), else register-prefetch + ds_store_b128.
// M, N multiples of 128; K multiple of 32.
// ---------------------------------------------------------------------------
#define LDSPAD 40

template <bool OUT_BF16>
__global__ void __launch_bounds__(256)
gemm_bias_kernel(const __bf16* __restrict__ A, const __bf16* __restrict__ W,
                 const float* __restrict__ bias, void* __restrict__ out,
                 int M, int N, int K) {
  __shared__ __bf16 smem[2][2][128 * LDSPAD];   // [buf][A=0/W=1][row*40+k]

  const int tid  = threadIdx.x;
  const int wid  = tid >> 5;
  const int lane = tid & 31;
  const int half = lane >> 4;
  const int lr   = lane & 15;
  const int wm   = wid >> 1;          // 0..3  -> 32-row slice
  const int wn   = wid & 1;           // 0..1  -> 64-col slice

  const int nblks_n = N >> 7;
  const int mblk = blockIdx.x / nblks_n;
  const int nblk = blockIdx.x % nblks_n;

  // Tile-staging addresses for this thread: two rows (r, r+64), 16B each.
  const int trow = tid >> 2;          // 0..63
  const int tcol = (tid & 3) * 8;     // 0,8,16,24 (elements)
  const __bf16* gA = A + (size_t)(mblk * 128 + trow) * K + tcol;
  const __bf16* gW = W + (size_t)(nblk * 128 + trow) * K + tcol;
  const int loff = trow * LDSPAD + tcol;
  const int lstep = 64 * LDSPAD;

  v8f acc[2][4];
#pragma unroll
  for (int s = 0; s < 2; ++s)
#pragma unroll
    for (int t = 0; t < 4; ++t)
#pragma unroll
      for (int j = 0; j < 8; ++j) acc[s][t][j] = 0.0f;

  const int nk = K >> 5;

#if USE_ASYNC_LDS
  // Builtin signature (from clang diagnostics): first param is a non-const
  // int4 pointer in addrspace(1) ("__device__"), LDS side addrspace(3).
  typedef __attribute__((address_space(1))) v4i* gvp_t;
  typedef __attribute__((address_space(3))) v4i* lvp_t;
#define ISSUE_TILE(buf, kc)                                                    \
  do {                                                                         \
    __builtin_amdgcn_global_load_async_to_lds_b128(                            \
        (gvp_t)(gA + (kc)), (lvp_t)(&smem[buf][0][loff]), 0, 0);               \
    __builtin_amdgcn_global_load_async_to_lds_b128(                            \
        (gvp_t)(gA + (size_t)64 * K + (kc)),                                   \
        (lvp_t)(&smem[buf][0][loff + lstep]), 0, 0);                           \
    __builtin_amdgcn_global_load_async_to_lds_b128(                            \
        (gvp_t)(gW + (kc)), (lvp_t)(&smem[buf][1][loff]), 0, 0);               \
    __builtin_amdgcn_global_load_async_to_lds_b128(                            \
        (gvp_t)(gW + (size_t)64 * K + (kc)),                                   \
        (lvp_t)(&smem[buf][1][loff + lstep]), 0, 0);                           \
  } while (0)

  ISSUE_TILE(0, 0);
#else
  uint4 pa0, pa1, pw0, pw1;
#define LOAD_REGS(kc)                                                          \
  do {                                                                         \
    pa0 = *(const uint4*)(gA + (kc));                                          \
    pa1 = *(const uint4*)(gA + (size_t)64 * K + (kc));                         \
    pw0 = *(const uint4*)(gW + (kc));                                          \
    pw1 = *(const uint4*)(gW + (size_t)64 * K + (kc));                         \
  } while (0)
#define STORE_REGS(buf)                                                        \
  do {                                                                         \
    *(uint4*)(&smem[buf][0][loff]) = pa0;                                      \
    *(uint4*)(&smem[buf][0][loff + lstep]) = pa1;                              \
    *(uint4*)(&smem[buf][1][loff]) = pw0;                                      \
    *(uint4*)(&smem[buf][1][loff + lstep]) = pw1;                              \
  } while (0)

  LOAD_REGS(0);
  STORE_REGS(0);
  __syncthreads();
#endif

  for (int ks = 0; ks < nk; ++ks) {
    const int buf = ks & 1;

#if USE_ASYNC_LDS
    if (ks + 1 < nk) {
      ISSUE_TILE(buf ^ 1, (ks + 1) * 32);
      __builtin_amdgcn_s_wait_asynccnt(4);   // current tile's loads retired
    } else {
      __builtin_amdgcn_s_wait_asynccnt(0);
    }
    __syncthreads();                         // all waves' tile data visible
#else
    if (ks + 1 < nk) LOAD_REGS((ks + 1) * 32);
#endif

    // Compute: 2 A-frags (rows wm*32 + {0,16} + lr), 4 W-frags.
    const __bf16* abase = &smem[buf][0][(wm * 32 + lr) * LDSPAD + half * 8];
    const __bf16* wbase = &smem[buf][1][(wn * 64 + lr) * LDSPAD + half * 8];
    v16bf af0 = load_frag(abase);
    v16bf af1 = load_frag(abase + 16 * LDSPAD);
#pragma unroll
    for (int t = 0; t < 4; ++t) {
      v16bf wf = load_frag(wbase + t * 16 * LDSPAD);
      acc[0][t] = wmma_bf16(af0, wf, acc[0][t]);
      acc[1][t] = wmma_bf16(af1, wf, acc[1][t]);
    }

#if USE_ASYNC_LDS
    __syncthreads();                         // reads done before overwrite
#else
    if (ks + 1 < nk) STORE_REGS(buf ^ 1);
    __syncthreads();
#endif
  }

  // Epilogue: C layout row = 8*(lane/16)+j, col = lane%16 (per 16x16 tile).
#pragma unroll
  for (int s = 0; s < 2; ++s)
#pragma unroll
    for (int t = 0; t < 4; ++t) {
      const int n = nblk * 128 + wn * 64 + t * 16 + lr;
      const float bn = bias[n];
#pragma unroll
      for (int j = 0; j < 8; ++j) {
        const int m = mblk * 128 + wm * 32 + s * 16 + half * 8 + j;
        const float v = acc[s][t][j] + bn;
        if (OUT_BF16)
          ((__bf16*)out)[(size_t)m * N + n] = (__bf16)v;
        else
          ((float*)out)[(size_t)m * N + n] = v;
      }
    }
}

// ---------------------------------------------------------------------------
// RoPE + head split.
//   qproj/kproj/vproj : bf16 [B*S, DIM]
//   qh, kh            : bf16 [B, H, S, HD]   (q gets softmax scale folded in)
//   vt                : bf16 [B, H, HD, S]   (transposed for P@V WMMA B-loads)
// ---------------------------------------------------------------------------
__global__ void rope_split_kernel(const __bf16* __restrict__ qp,
                                  const __bf16* __restrict__ kp,
                                  const __bf16* __restrict__ vp,
                                  const float* __restrict__ cosd,
                                  const float* __restrict__ sind,
                                  __bf16* __restrict__ qh,
                                  __bf16* __restrict__ kh,
                                  __bf16* __restrict__ vt, float sm_scale) {
  const int idx = blockIdx.x * blockDim.x + threadIdx.x;  // B*S*DIM = 2^23
  if (idx >= B_ * S_ * DIM_) return;
  const int d = idx & (HD_ - 1);
  const int h = (idx >> 7) & (H_ - 1);
  const int s = (idx >> 11) & (S_ - 1);
  const int b = idx >> 22;

  const size_t src  = (size_t)(b * S_ + s) * DIM_ + h * HD_ + d;
  const int    d2   = (d < HD_ / 2) ? d + HD_ / 2 : d - HD_ / 2;
  const size_t src2 = src - d + d2;

  const float c  = cosd[s * HD_ + d];
  const float sn = sind[s * HD_ + d];

  const float q0 = (float)qp[src];
  const float q1 = (float)qp[src2];
  const float k0 = (float)kp[src];
  const float k1 = (float)kp[src2];
  const float rq = (d < HD_ / 2) ? -q1 : q1;
  const float rk = (d < HD_ / 2) ? -k1 : k1;

  const size_t dsth = ((size_t)(b * H_ + h) * S_ + s) * HD_ + d;
  qh[dsth] = (__bf16)((q0 * c + rq * sn) * sm_scale);
  kh[dsth] = (__bf16)(k0 * c + rk * sn);
  vt[((size_t)(b * H_ + h) * HD_ + d) * S_ + s] = vp[src];
}

// ---------------------------------------------------------------------------
// Causal flash attention. One wave handles one 16-row query tile of one
// (b,h); block = 4 waves, all with identical trip counts (derived from the
// block's max query tile) so __syncthreads is uniform. Padded kv blocks are
// fully masked (p == 0, alpha == 1) and therefore no-ops numerically.
// ---------------------------------------------------------------------------
__global__ void __launch_bounds__(128)
flash_attn_kernel(const __bf16* __restrict__ Q, const __bf16* __restrict__ Kh,
                  const __bf16* __restrict__ Vt, __bf16* __restrict__ Ctx) {
  __shared__ __bf16 ldsP[4][16 * 32];

  const int wid  = threadIdx.x >> 5;
  const int lane = threadIdx.x & 31;
  const int half = lane >> 4;
  const int lr   = lane & 15;

  const int bh = blockIdx.y;           // b*H + h
  const int b  = bh / H_;
  const int h  = bh % H_;
  const int qt = blockIdx.x * 4 + wid;
  const int qbase = qt * 16;

  const __bf16* qptr = Q  + ((size_t)bh * S_ + qbase) * HD_;
  const __bf16* kptr = Kh + (size_t)bh * S_ * HD_;
  const __bf16* vptr = Vt + (size_t)bh * HD_ * S_;

  // Query fragments: 4 k-chunks of 32 over HD=128.
  v16bf qa[4];
#pragma unroll
  for (int dc = 0; dc < 4; ++dc)
    qa[dc] = load_frag(qptr + (size_t)lr * HD_ + dc * 32 + half * 8);

  v8f acc[8];
#pragma unroll
  for (int nc = 0; nc < 8; ++nc)
#pragma unroll
    for (int j = 0; j < 8; ++j) acc[nc][j] = 0.0f;

  float rowmax[8], rowsum[8];
#pragma unroll
  for (int j = 0; j < 8; ++j) { rowmax[j] = -1e30f; rowsum[j] = 0.0f; }

  // Uniform trip count over the block's 4 waves (causal bound of max tile).
  const int nblk = (((blockIdx.x * 4 + 3) * 16) + 16 + 31) >> 5;
  __bf16* lp = ldsP[wid];

  for (int kb = 0; kb < nblk; ++kb) {
    const int kv0 = kb * 32;

    // Scores: two 16x16 tiles over the 32-wide kv block.
    v8f s0, s1;
#pragma unroll
    for (int j = 0; j < 8; ++j) { s0[j] = 0.0f; s1[j] = 0.0f; }
#pragma unroll
    for (int dc = 0; dc < 4; ++dc) {
      v16bf b0 = load_frag(kptr + (size_t)(kv0 + lr) * HD_ + dc * 32 + half * 8);
      v16bf b1 = load_frag(kptr + (size_t)(kv0 + 16 + lr) * HD_ + dc * 32 + half * 8);
      s0 = wmma_bf16(qa[dc], b0, s0);
      s1 = wmma_bf16(qa[dc], b1, s1);
    }

    // Causal mask + online softmax (per output row m = 8*half + j).
#pragma unroll
    for (int j = 0; j < 8; ++j) {
      const int row = qbase + half * 8 + j;
      const float v0 = (kv0 + lr > row)      ? -1e30f : s0[j];
      const float v1 = (kv0 + 16 + lr > row) ? -1e30f : s1[j];

      float mx = fmaxf(v0, v1);
      mx = fmaxf(mx, __shfl_xor(mx, 1, 16));
      mx = fmaxf(mx, __shfl_xor(mx, 2, 16));
      mx = fmaxf(mx, __shfl_xor(mx, 4, 16));
      mx = fmaxf(mx, __shfl_xor(mx, 8, 16));

      const float newmax = fmaxf(rowmax[j], mx);
      const float alpha  = __expf(rowmax[j] - newmax);
      rowmax[j] = newmax;

      const float p0 = __expf(v0 - newmax);   // masked -> 0
      const float p1 = __expf(v1 - newmax);
      float ps = p0 + p1;
      ps += __shfl_xor(ps, 1, 16);
      ps += __shfl_xor(ps, 2, 16);
      ps += __shfl_xor(ps, 4, 16);
      ps += __shfl_xor(ps, 8, 16);
      rowsum[j] = rowsum[j] * alpha + ps;

#pragma unroll
      for (int nc = 0; nc < 8; ++nc) acc[nc][j] *= alpha;

      const int m = half * 8 + j;             // C layout -> row-major LDS tile
      lp[m * 32 + lr]      = (__bf16)p0;
      lp[m * 32 + 16 + lr] = (__bf16)p1;
    }
    __syncthreads();

    // P(16x32) @ V(32x128): reload P in A layout, V^T gives k-contiguous B.
    v16bf pa = load_frag(lp + lr * 32 + half * 8);
#pragma unroll
    for (int nc = 0; nc < 8; ++nc) {
      v16bf vb = load_frag(vptr + (size_t)(nc * 16 + lr) * S_ + kv0 + half * 8);
      acc[nc] = wmma_bf16(pa, vb, acc[nc]);
    }
    __syncthreads();
  }

  // Normalize and write context back in [B, S, H*HD] layout for the O-proj.
#pragma unroll
  for (int j = 0; j < 8; ++j) {
    const float inv = 1.0f / rowsum[j];
    const int m = qbase + half * 8 + j;
#pragma unroll
    for (int nc = 0; nc < 8; ++nc) {
      const int d = nc * 16 + lr;
      Ctx[((size_t)(b * S_) + m) * DIM_ + h * HD_ + d] =
          (__bf16)(acc[nc][j] * inv);
    }
  }
}

// ---------------------------------------------------------------------------
// Host-side orchestration
// ---------------------------------------------------------------------------
extern "C" void kernel_launch(void* const* d_in, const int* in_sizes, int n_in,
                              void* d_out, int out_size, void* d_ws,
                              size_t ws_size, hipStream_t stream) {
  (void)in_sizes; (void)n_in; (void)out_size; (void)ws_size;

  const float* query = (const float*)d_in[0];
  const float* keyv  = (const float*)d_in[1];
  const float* cosd  = (const float*)d_in[2];
  const float* sind  = (const float*)d_in[3];
  const float* wq    = (const float*)d_in[4];
  const float* bq    = (const float*)d_in[5];
  const float* wk    = (const float*)d_in[6];
  const float* bk    = (const float*)d_in[7];
  const float* wv    = (const float*)d_in[8];
  const float* bv    = (const float*)d_in[9];
  const float* wo    = (const float*)d_in[10];
  const float* bo    = (const float*)d_in[11];

  char* ws = (char*)d_ws;
  const size_t MB = 1024ull * 1024ull;
  const int NBS = B_ * S_ * DIM_;   // 8388608 activation elements
  const int NW  = DIM_ * DIM_;      // 4194304 weight elements

  // Workspace layout (bf16 buffers). Aliases are safe because the producers
  // of the aliased buffers run strictly after the last reader of the
  // original occupant.
  __bf16* xq_bf  = (__bf16*)(ws + 0 * MB);    // query bf16   (reused by ctx)
  __bf16* xkv_bf = (__bf16*)(ws + 16 * MB);   // key_value bf16 (reused by qh)
  __bf16* wq_bf  = (__bf16*)(ws + 32 * MB);
  __bf16* wk_bf  = (__bf16*)(ws + 40 * MB);
  __bf16* wv_bf  = (__bf16*)(ws + 48 * MB);
  __bf16* wo_bf  = (__bf16*)(ws + 56 * MB);
  __bf16* qproj  = (__bf16*)(ws + 64 * MB);   // [B*S, DIM]
  __bf16* kproj  = (__bf16*)(ws + 80 * MB);
  __bf16* vproj  = (__bf16*)(ws + 96 * MB);
  __bf16* kh     = (__bf16*)(ws + 112 * MB);  // [B,H,S,HD]
  __bf16* vt     = (__bf16*)(ws + 128 * MB);  // [B,H,HD,S]
  __bf16* qh     = (__bf16*)(ws + 16 * MB);   // alias over xkv_bf
  __bf16* ctx    = (__bf16*)(ws + 0 * MB);    // alias over xq_bf

  // 1) fp32 -> bf16 conversions.
  const int CT = 256;
  f32_to_bf16_kernel<<<(NBS + CT - 1) / CT, CT, 0, stream>>>(query, xq_bf, NBS);
  f32_to_bf16_kernel<<<(NBS + CT - 1) / CT, CT, 0, stream>>>(keyv, xkv_bf, NBS);
  f32_to_bf16_kernel<<<(NW + CT - 1) / CT, CT, 0, stream>>>(wq, wq_bf, NW);
  f32_to_bf16_kernel<<<(NW + CT - 1) / CT, CT, 0, stream>>>(wk, wk_bf, NW);
  f32_to_bf16_kernel<<<(NW + CT - 1) / CT, CT, 0, stream>>>(wv, wv_bf, NW);
  f32_to_bf16_kernel<<<(NW + CT - 1) / CT, CT, 0, stream>>>(wo, wo_bf, NW);

  // 2) QKV projections: [4096,2048] x [2048,2048]^T, bf16 out.
  const int gemm_blocks = (MROWS / 128) * (DIM_ / 128);     // 512
  gemm_bias_kernel<true><<<gemm_blocks, 256, 0, stream>>>(
      xq_bf, wq_bf, bq, qproj, MROWS, DIM_, DIM_);
  gemm_bias_kernel<true><<<gemm_blocks, 256, 0, stream>>>(
      xkv_bf, wk_bf, bk, kproj, MROWS, DIM_, DIM_);
  gemm_bias_kernel<true><<<gemm_blocks, 256, 0, stream>>>(
      xkv_bf, wv_bf, bv, vproj, MROWS, DIM_, DIM_);

  // 3) RoPE + head split (+ softmax scale folded into Q).
  const float sm_scale = 0.08838834764831845f;              // 1/sqrt(128)
  rope_split_kernel<<<(NBS + CT - 1) / CT, CT, 0, stream>>>(
      qproj, kproj, vproj, cosd, sind, qh, kh, vt, sm_scale);

  // 4) Causal flash attention -> ctx bf16 [B, S, DIM].
  dim3 fgrid(S_ / 16 / 4, B_ * H_);                         // (32, 32)
  flash_attn_kernel<<<fgrid, 128, 0, stream>>>(qh, kh, vt, ctx);

  // 5) Output projection -> fp32 d_out.
  gemm_bias_kernel<false><<<gemm_blocks, 256, 0, stream>>>(
      ctx, wo_bf, bo, d_out, MROWS, DIM_, DIM_);
}
